// Attention_65575560675510
// MI455X (gfx1250) — matrix operand order repaired and checked by reading, not compiled, not run on hardware
//
#include <hip/hip_runtime.h>
#include <hip/hip_bf16.h>

// ---------------------------------------------------------------------------
// DINOv3-style attention block on gfx1250 (MI455X), wave32 + WMMA bf16.
//   B=2, H=W=48 -> N=2304 tokens, C=768, heads=12, head_dim=64.
// Pipeline:
//   k0: pre-convert w_qkv / w_proj to bf16 (once) in workspace
//   k1: qkv = x @ w_qkv^T + b  (WMMA bf16), fused in-register RoPE on q/k,
//       writes Q[b,h,n,64], K[b,h,n,64], V^T[b,h,64,n]  (all bf16, in d_ws)
//   k2: flash-style attention; K/V tiles staged into LDS with double-buffered
//       global_load_async_to_lds_b128 (ASYNCcnt) shared by all 4 waves of the
//       block; WMMA bf16 for QK^T and PV; online softmax via 16-lane
//       shfl_xor reductions + v_exp_f32; writes O[b,n,768] bf16
//   k3: out = O @ w_proj^T + b_proj (WMMA bf16), f32 output to d_out
// ---------------------------------------------------------------------------

typedef __attribute__((ext_vector_type(16))) __bf16 v16bf;
typedef __attribute__((ext_vector_type(8)))  __bf16 v8bf;
typedef __attribute__((ext_vector_type(8)))  float  v8f;

#define DIMC   768
#define NTOK   2304
#define NHEAD  12
#define HD     64
#define SIDE   48

__device__ __forceinline__ v8f wmma_bf16(v16bf a, v16bf b, v8f c) {
  return __builtin_amdgcn_wmma_f32_16x16x32_bf16(false, a, false, b,
                                                 (short)0, c, false, false);
}

// Async DMA: 16 bytes global -> LDS, tracked by ASYNCcnt (ISA 08, §4).
// LDS address = low 32 bits of the flat shared-aperture address.
__device__ __forceinline__ void async_ld16(void* lds_ptr, const void* gptr) {
  unsigned lds_off = (unsigned)(size_t)lds_ptr;
  unsigned long long ga = (unsigned long long)(size_t)gptr;
  asm volatile("global_load_async_to_lds_b128 %0, %1, off"
               :: "v"(lds_off), "v"(ga)
               : "memory");
}
__device__ __forceinline__ void wait_async0() {
  asm volatile("s_wait_asynccnt 0" ::: "memory");
}

// A-fragment (16x32 bf16) from an f32 row: lane&15 = M-row (caller passes the
// row pointer), lane>>4 = K-split half:
//   lanes 0-15 : K = {kb+0..15};  lanes 16-31: K = {kb+16..31}
__device__ __forceinline__ v16bf a_frag_f32(const float* rowp, int kb, int hi) {
  union { v16bf v; __bf16 e[16]; } u;
  const float* p0 = rowp + kb + hi * 16;
  const float* p1 = p0 + 8;
#pragma unroll
  for (int i = 0; i < 8; ++i) u.e[i] = (__bf16)p0[i];
#pragma unroll
  for (int i = 0; i < 8; ++i) u.e[8 + i] = (__bf16)p1[i];
  return u.v;
}

// A-fragment from a bf16 row (same K-split).
__device__ __forceinline__ v16bf a_frag_bf16(const __bf16* rowp, int kb, int hi) {
  union { v16bf v; v8bf h2[2]; } u;
  const __bf16* p = rowp + kb + hi * 16;
  u.h2[0] = *(const v8bf*)(p);
  u.h2[1] = *(const v8bf*)(p + 8);
  return u.v;
}

// B-fragment (32x16) from bf16 weights W[col, k]: lane&15 = column n (caller
// passes col), lane>>4 picks 16 contiguous k — one aligned 32-byte load.
__device__ __forceinline__ v16bf b_frag_bf16(const __bf16* W, int ldw, int col,
                                             int kb, int hi) {
  return *(const v16bf*)(W + (size_t)col * ldw + kb + hi * 16);
}

// ---------------------------------------------------------------------------
// Kernel 0: one-time f32 -> bf16 weight conversion.
// ---------------------------------------------------------------------------
__global__ __launch_bounds__(256) void cvt_w_kernel(
    const float* __restrict__ wq, const float* __restrict__ wp,
    __bf16* __restrict__ wqb, __bf16* __restrict__ wpb) {
  const int i = blockIdx.x * 256 + threadIdx.x;
  if (i < 3 * DIMC * DIMC) wqb[i] = (__bf16)wq[i];
  if (i < DIMC * DIMC) wpb[i] = (__bf16)wp[i];
}

// ---------------------------------------------------------------------------
// Kernel 1: QKV projection + bias + fused RoPE, bf16 outputs.
// grid (288, 9), block 128 (4 waves).  wave -> one 16-row x 64-col tile.
// ---------------------------------------------------------------------------
__global__ __launch_bounds__(128) void qkv_rope_kernel(
    const float* __restrict__ x, const __bf16* __restrict__ Wq,
    const float* __restrict__ b_qkv, __bf16* __restrict__ Qw,
    __bf16* __restrict__ Kw, __bf16* __restrict__ Vt) {
  const int lane = threadIdx.x & 31;
  const int wave = threadIdx.x >> 5;
  const int n  = lane & 15;
  const int hi = lane >> 4;
  const int mtile = blockIdx.x;            // 0..287  (16 rows each, B*N/16)
  const int cg    = blockIdx.y * 4 + wave; // 0..35   (64 columns each)
  const int r0 = mtile * 16;
  const int j0 = cg * 64;

  const float* xrow = x + (size_t)(r0 + n) * DIMC;

  v8f acc[4] = {};
  for (int kb = 0; kb < DIMC; kb += 32) {
    __builtin_prefetch(xrow + kb + 128, 0, 3);
    v16bf A = a_frag_f32(xrow, kb, hi);
#pragma unroll
    for (int t = 0; t < 4; ++t) {
      v16bf Bf = b_frag_bf16(Wq, DIMC, j0 + t * 16 + n, kb, hi);
      acc[t] = wmma_bf16(A, Bf, acc[t]);
    }
  }

#pragma unroll
  for (int t = 0; t < 4; ++t) {
    float bias = b_qkv[j0 + t * 16 + n];
#pragma unroll
    for (int i = 0; i < 8; ++i) acc[t][i] += bias;
  }

  const int group = cg / NHEAD;  // 0=q,1=k,2=v
  const int head  = cg % NHEAD;

  if (group < 2) {
    // In-register RoPE: feature pair (f, f+32) lives in tiles (t, t+2) at the
    // same lane/register; sin/cos are period-32 duplicated.
#pragma unroll
    for (int t = 0; t < 2; ++t) {
      const int f    = t * 16 + n;   // 0..31
      const int axis = f >> 4;       // 0: y-coord, 1: x-coord
      const int idx  = f & 15;
      const float period = __powf(100.0f, (float)idx * (1.0f / 16.0f));
#pragma unroll
      for (int i = 0; i < 8; ++i) {
        const int r   = r0 + i + 8 * hi;
        const int tok = r % NTOK;
        const int cc  = axis ? (tok % SIDE) : (tok / SIDE);
        const float cn  = 2.0f * (((float)cc + 0.5f) * (1.0f / SIDE)) - 1.0f;
        const float ang = 6.283185307179586f * cn / period;
        float s, c;
        __sincosf(ang, &s, &c);
        const float lo = acc[t][i], hv = acc[t + 2][i];
        acc[t][i]     = lo * c - hv * s;
        acc[t + 2][i] = hv * c + lo * s;
      }
    }
  }

  __bf16* dst = (group == 0) ? Qw : (group == 1) ? Kw : Vt;
#pragma unroll
  for (int t = 0; t < 4; ++t) {
    const int col = t * 16 + n;  // feature within head
#pragma unroll
    for (int i = 0; i < 8; ++i) {
      const int r   = r0 + i + 8 * hi;
      const int b   = r / NTOK;
      const int tok = r % NTOK;
      size_t off;
      if (group == 2)  // V stored transposed: [b,h,feature,token]
        off = (((size_t)(b * NHEAD + head) * HD + col) * NTOK + tok);
      else             // Q/K: [b,h,token,feature]
        off = (((size_t)(b * NHEAD + head) * NTOK + tok) * HD + col);
      dst[off] = (__bf16)acc[t][i];
    }
  }
}

// ---------------------------------------------------------------------------
// Kernel 2: flash-style attention with async-LDS K/V staging.
// grid (24, 36), block 128 (4 waves).  All 4 waves share one (b,h): the
// block double-buffers 32-key K/V tiles in LDS via ASYNC DMA; each wave
// owns 16 queries.
// ---------------------------------------------------------------------------
__global__ __launch_bounds__(128) void attn_kernel(
    const __bf16* __restrict__ Qw, const __bf16* __restrict__ Kw,
    const __bf16* __restrict__ Vt, __bf16* __restrict__ Ow) {
  __shared__ __align__(32) __bf16 Kbuf[2][32 * HD];  // [key_local][feature]
  __shared__ __align__(32) __bf16 Vbuf[2][HD * 32];  // [feature][key_local]
  __shared__ __align__(32) __bf16 Pst[4][16 * 32];   // per-wave P staging

  const int tid  = threadIdx.x;
  const int lane = tid & 31;
  const int wave = tid >> 5;
  const int n  = lane & 15;
  const int hi = lane >> 4;
  const int bh = blockIdx.x;                 // 0..23
  const int qt = blockIdx.y * 4 + wave;      // 0..143
  const int b = bh / NHEAD, h = bh % NHEAD;

  const __bf16* Qb = Qw + (size_t)bh * NTOK * HD;
  const __bf16* Kb = Kw + (size_t)bh * NTOK * HD;
  const __bf16* Vb = Vt + (size_t)bh * HD * NTOK;

  // Q A-fragments: 16 queries x 64 features -> two 16x32 fragments.
  const __bf16* qrow = Qb + (size_t)(qt * 16 + n) * HD;
  v16bf qf[2];
#pragma unroll
  for (int fr = 0; fr < 2; ++fr) qf[fr] = a_frag_bf16(qrow, fr * 32, hi);

  v8f o[4] = {};
  float mrow[8], lrow[8];
#pragma unroll
  for (int i = 0; i < 8; ++i) { mrow[i] = -3.0e38f; lrow[i] = 0.0f; }

  __bf16* lp = &Pst[wave][0];
  const float scale = 0.125f;  // 64^-0.5

  // Issue async copy of one 32-key tile (K: 4KB contiguous, V: 64 strided
  // rows of 64B).  256 x 16B chunks per tile pair half -> 2 chunks/thread.
  auto issue_tile = [&](int kt, int buf) {
    const __bf16* gk = Kb + (size_t)kt * HD;  // 2048 contiguous bf16
#pragma unroll
    for (int r = 0; r < 2; ++r) {
      const int c = tid + r * 128;            // 0..255
      async_ld16(&Kbuf[buf][c * 8], gk + c * 8);
      const int f = c >> 2, sub = c & 3;
      async_ld16(&Vbuf[buf][f * 32 + sub * 8],
                 Vb + (size_t)f * NTOK + kt + sub * 8);
    }
  };

  issue_tile(0, 0);

  for (int it = 0; it < NTOK / 32; ++it) {
    const int buf = it & 1;
    const int kt  = it * 32;
    wait_async0();        // this wave's DMA for tile `it` has landed in LDS
    __syncthreads();      // ... and everyone else's too
    if (it + 1 < NTOK / 32) issue_tile(kt + 32, buf ^ 1);  // overlap next DMA

    const __bf16* kl = &Kbuf[buf][0];
    const __bf16* vl = &Vbuf[buf][0];

    // --- scores: two 16-key tiles, K-dim 64 = 2 WMMA steps each ---
    v8f s0 = {}, s1 = {};
#pragma unroll
    for (int fr = 0; fr < 2; ++fr) {
      v16bf kf0 = *(const v16bf*)(kl + (size_t)n * HD + fr * 32 + hi * 16);
      v16bf kf1 = *(const v16bf*)(kl + (size_t)(16 + n) * HD + fr * 32 + hi * 16);
      s0 = wmma_bf16(qf[fr], kf0, s0);
      s1 = wmma_bf16(qf[fr], kf1, s1);
    }

    // --- online softmax (row stats across the 16-lane column group) ---
    float p0[8], p1[8], alpha[8];
#pragma unroll
    for (int i = 0; i < 8; ++i) {
      float a0 = s0[i] * scale, a1 = s1[i] * scale;
      float mx = fmaxf(a0, a1);
      mx = fmaxf(mx, __shfl_xor(mx, 1, 32));
      mx = fmaxf(mx, __shfl_xor(mx, 2, 32));
      mx = fmaxf(mx, __shfl_xor(mx, 4, 32));
      mx = fmaxf(mx, __shfl_xor(mx, 8, 32));
      const float mn = fmaxf(mrow[i], mx);
      alpha[i] = __expf(mrow[i] - mn);
      p0[i] = __expf(a0 - mn);
      p1[i] = __expf(a1 - mn);
      float sum = p0[i] + p1[i];
      sum += __shfl_xor(sum, 1, 32);
      sum += __shfl_xor(sum, 2, 32);
      sum += __shfl_xor(sum, 4, 32);
      sum += __shfl_xor(sum, 8, 32);
      lrow[i] = lrow[i] * alpha[i] + sum;
      mrow[i] = mn;
    }
#pragma unroll
    for (int t = 0; t < 4; ++t)
#pragma unroll
      for (int i = 0; i < 8; ++i) o[t][i] *= alpha[i];

    // --- stage P (D-layout) through LDS into A-fragment layout ---
#pragma unroll
    for (int i = 0; i < 8; ++i) {
      const int m = i + 8 * hi;
      lp[m * 32 + n]      = (__bf16)p0[i];
      lp[m * 32 + 16 + n] = (__bf16)p1[i];
    }
    v16bf pa = a_frag_bf16(lp + n * 32, 0, hi);

    // --- PV: 4 feature tiles, B-fragments from LDS-resident V^T ---
#pragma unroll
    for (int t = 0; t < 4; ++t) {
      v16bf vf = *(const v16bf*)(vl + (size_t)(t * 16 + n) * 32 + hi * 16);
      o[t] = wmma_bf16(pa, vf, o[t]);
    }
  }

  // normalize + store O[b, tok, h*64 + f] (bf16 for the projection GEMM)
#pragma unroll
  for (int t = 0; t < 4; ++t)
#pragma unroll
    for (int i = 0; i < 8; ++i) {
      const int tok = qt * 16 + i + 8 * hi;
      const float ov = o[t][i] / lrow[i];
      Ow[((size_t)b * NTOK + tok) * DIMC + h * HD + t * 16 + n] = (__bf16)ov;
    }
}

// ---------------------------------------------------------------------------
// Kernel 3: output projection.  grid (288, 3), block 128 (4 waves).
// ---------------------------------------------------------------------------
__global__ __launch_bounds__(128) void proj_kernel(
    const __bf16* __restrict__ Ow, const __bf16* __restrict__ Wp,
    const float* __restrict__ b_proj, float* __restrict__ out) {
  const int lane = threadIdx.x & 31;
  const int wave = threadIdx.x >> 5;
  const int n  = lane & 15;
  const int hi = lane >> 4;
  const int mtile = blockIdx.x;            // 0..287
  const int cg    = blockIdx.y * 4 + wave; // 0..11
  const int r0 = mtile * 16;
  const int j0 = cg * 64;

  const __bf16* arow = Ow + (size_t)(r0 + n) * DIMC;

  v8f acc[4] = {};
  for (int kb = 0; kb < DIMC; kb += 32) {
    __builtin_prefetch(arow + kb + 128, 0, 3);
    v16bf A = a_frag_bf16(arow, kb, hi);
#pragma unroll
    for (int t = 0; t < 4; ++t) {
      v16bf Bf = b_frag_bf16(Wp, DIMC, j0 + t * 16 + n, kb, hi);
      acc[t] = wmma_bf16(A, Bf, acc[t]);
    }
  }

#pragma unroll
  for (int t = 0; t < 4; ++t) {
    const float bias = b_proj[j0 + t * 16 + n];
#pragma unroll
    for (int i = 0; i < 8; ++i) {
      const int r = r0 + i + 8 * hi;
      out[(size_t)r * DIMC + j0 + t * 16 + n] = acc[t][i] + bias;
    }
  }
}

// ---------------------------------------------------------------------------
extern "C" void kernel_launch(void* const* d_in, const int* in_sizes, int n_in,
                              void* d_out, int out_size, void* d_ws,
                              size_t ws_size, hipStream_t stream) {
  (void)in_sizes; (void)n_in; (void)out_size; (void)ws_size;
  const float* x      = (const float*)d_in[0];
  const float* w_qkv  = (const float*)d_in[1];
  const float* b_qkv  = (const float*)d_in[2];
  const float* w_proj = (const float*)d_in[3];
  const float* b_proj = (const float*)d_in[4];
  float* out = (float*)d_out;

  // workspace partition (bf16): Q, K, V^T, O, Wq_bf16, Wp_bf16
  const size_t SEG = (size_t)2 * NHEAD * NTOK * HD;  // 3,538,944 elements
  __bf16* Qw  = (__bf16*)d_ws;
  __bf16* Kw  = Qw + SEG;
  __bf16* Vt  = Kw + SEG;
  __bf16* Ow  = Vt + SEG;
  __bf16* Wqb = Ow + SEG;                       // 3*768*768
  __bf16* Wpb = Wqb + (size_t)3 * DIMC * DIMC;  // 768*768

  dim3 blk(128);
  cvt_w_kernel<<<dim3((3 * DIMC * DIMC + 255) / 256), dim3(256), 0, stream>>>(
      w_qkv, w_proj, Wqb, Wpb);
  qkv_rope_kernel<<<dim3(288, 9), blk, 0, stream>>>(x, Wqb, b_qkv, Qw, Kw, Vt);
  attn_kernel<<<dim3(24, 36), blk, 0, stream>>>(Qw, Kw, Vt, Ow);
  proj_kernel<<<dim3(288, 3), blk, 0, stream>>>(Ow, Wpb, b_proj, out);
}
